// QuantResnetBlockINT4_9706626089383
// MI455X (gfx1250) — compile-verified, matrix operand order checked by reading, not run on hardware
//
#include <hip/hip_runtime.h>
#include <hip/hip_bf16.h>
#include <math.h>

typedef __attribute__((ext_vector_type(16))) __bf16 v16bf;
typedef __attribute__((ext_vector_type(8)))  float  v8f;
typedef unsigned int v4u __attribute__((ext_vector_type(4)));
typedef unsigned int v8u __attribute__((ext_vector_type(8)));

#define Bsz  16
#define CIN  256
#define COUT 512
#define HW   4096

// ------------------------------------------------------------------
// Weight repack: [Cout][Cin][3][3] f32 -> [Cout][K] bf16, k = p*Cin + c
// ------------------------------------------------------------------
__global__ void pack_w3_kernel(const float* __restrict__ w, __bf16* __restrict__ wp, int Cin) {
  int idx = blockIdx.x * blockDim.x + threadIdx.x;
  int total = COUT * Cin * 9;
  if (idx >= total) return;
  int p  = idx % 9;
  int c  = (idx / 9) % Cin;
  int co = idx / (9 * Cin);
  wp[co * (Cin * 9) + p * Cin + c] = (__bf16)w[idx];
}

__global__ void pack_w1_kernel(const float* __restrict__ w, __bf16* __restrict__ wp) {
  int idx = blockIdx.x * blockDim.x + threadIdx.x;
  if (idx >= COUT * CIN) return;
  wp[idx] = (__bf16)w[idx];
}

// ------------------------------------------------------------------
// temb projection: tproj[b][co] = sum_t silu(temb[b][t]) * tw[co][t] + tb[co]
// ------------------------------------------------------------------
__global__ void temb_proj_kernel(const float* __restrict__ temb, const float* __restrict__ tw,
                                 const float* __restrict__ tb, float* __restrict__ tproj) {
  int idx = blockIdx.x * blockDim.x + threadIdx.x;  // b*COUT + co
  if (idx >= Bsz * COUT) return;
  int co = idx % COUT;
  int b  = idx / COUT;
  const float* tr = temb + b * 512;
  const float* wr = tw + co * 512;
  float acc = tb[co];
  for (int t = 0; t < 512; ++t) {
    float v = tr[t];
    acc += (v / (1.f + __expf(-v))) * wr[t];
  }
  tproj[idx] = acc;
}

// ------------------------------------------------------------------
// GroupNorm stats: one block per (b, group); base = bg * Cg * HW.
// ------------------------------------------------------------------
__global__ void gn_stats_kernel(const float* __restrict__ x, float* __restrict__ mean,
                                float* __restrict__ rstd, int Cg) {
  int bg = blockIdx.x;
  long base = (long)bg * Cg * HW;
  int n = Cg * HW;
  float s = 0.f, s2 = 0.f;
  for (int i = threadIdx.x; i < n; i += 256) {
    float v = x[base + i];
    s += v; s2 += v * v;
  }
  __shared__ float sh1[256], sh2[256];
  sh1[threadIdx.x] = s; sh2[threadIdx.x] = s2;
  __syncthreads();
  for (int off = 128; off > 0; off >>= 1) {
    if (threadIdx.x < off) {
      sh1[threadIdx.x] += sh1[threadIdx.x + off];
      sh2[threadIdx.x] += sh2[threadIdx.x + off];
    }
    __syncthreads();
  }
  if (threadIdx.x == 0) {
    float m   = sh1[0] / (float)n;
    float var = sh2[0] / (float)n - m * m;
    mean[bg] = m;
    rstd[bg] = rsqrtf(var + 1e-6f);
  }
}

// GN1 + SiLU -> bf16, plus plain bf16 copy of x for the 1x1 shortcut GEMM
__global__ void gn1_apply_kernel(const float* __restrict__ x, const float* __restrict__ mean,
                                 const float* __restrict__ rstd, const float* __restrict__ gw,
                                 const float* __restrict__ gb, __bf16* __restrict__ h1b,
                                 __bf16* __restrict__ xb) {
  int idx = blockIdx.x * 256 + threadIdx.x;      // B*CIN*HW = 2^24 total
  int c  = (idx >> 12) & 255;
  int b  = idx >> 20;
  int bg = b * 32 + (c >> 3);                    // Cg = 8
  float v = x[idx];
  float nrm = (v - mean[bg]) * rstd[bg] * gw[c] + gb[c];
  h1b[idx] = (__bf16)(nrm / (1.f + __expf(-nrm)));
  xb[idx]  = (__bf16)v;
}

// GN2 + SiLU -> bf16
__global__ void gn2_apply_kernel(const float* __restrict__ h2, const float* __restrict__ mean,
                                 const float* __restrict__ rstd, const float* __restrict__ gw,
                                 const float* __restrict__ gb, __bf16* __restrict__ h2s) {
  int idx = blockIdx.x * 256 + threadIdx.x;      // B*COUT*HW = 2^25 total
  int c  = (idx >> 12) & 511;
  int b  = idx >> 21;
  int bg = b * 32 + (c >> 4);                    // Cg = 16
  float v = h2[idx];
  float nrm = (v - mean[bg]) * rstd[bg] * gw[c] + gb[c];
  h2s[idx] = (__bf16)(nrm / (1.f + __expf(-nrm)));
}

// ------------------------------------------------------------------
// TDM: load a [tile_dim1=128 rows x tile_dim0=32] bf16 tile from a row-major
// [512][K] matrix into LDS, HW row padding (16 DWORDs data + 4 DWORDs pad ->
// LDK=40 bf16 rows).  D# per ISA ch.8; 2-group form for a 2D tensor.
// ------------------------------------------------------------------
__device__ __forceinline__ void tdm_load_A(const __bf16* gsrc, unsigned lds_byte, int K) {
  unsigned long long ga = (unsigned long long)(size_t)gsrc;
  v4u g0 = {
      1u,                                        // count=1, user desc
      lds_byte,                                  // lds_addr
      (unsigned)ga,                              // global_addr[31:0]
      (unsigned)(ga >> 32) | (2u << 30)          // global_addr[56:32] | type=2
  };
  v8u g1 = {
      (1u << 16) | (1u << 20) | (3u << 22) | (3u << 25),
      //  data_size=2B | pad_enable | pad_interval=16dw | pad_amount=4dw
      ((unsigned)K & 0xFFFFu) << 16,             // tensor_dim0[15:0] @ [63:48]
      ((unsigned)K >> 16) | (512u << 16),        // tensor_dim0[31:16] | tensor_dim1[15:0]
      (32u << 16),                               // tensor_dim1[31:16]=0 | tile_dim0=32
      128u,                                      // tile_dim1=128 | tile_dim2=0
      (unsigned)K,                               // tensor_dim0_stride[31:0] (elements)
      0u,                                        // stride[47:32]=0 | dim1_stride lo=0
      0u
  };
  asm volatile("tensor_load_to_lds %0, %1" :: "s"(g0), "s"(g1) : "memory");
}

__device__ __forceinline__ void tdm_wait0() {
#if __has_builtin(__builtin_amdgcn_s_wait_tensorcnt)
  __builtin_amdgcn_s_wait_tensorcnt(0);
#else
  asm volatile("s_wait_tensorcnt 0x0" ::: "memory");
#endif
}

// ------------------------------------------------------------------
// Implicit-GEMM conv via WMMA bf16 (16x16x32, f32 accum), double-buffered LDS,
// A-tile staged by the Tensor Data Mover (wave 0), B-tile im2col gather.
// Block tile 128(M) x 256(N), 8 waves in 2x4, wave tile 64x64 -> 16 WMMAs and
// 16 ds_load_b128 fragment loads per wave per k-step.
// K packed as k = p*Cin + c; k-step t: p = t >> ncsh, c0 = (t & (nc-1))*32.
// mode 0: out = acc + bias + tproj   (conv1 -> h2 fp32)
// mode 1: out = acc + bias           (nin   -> d_out)
// mode 2: out += acc + bias          (conv2 -> d_out, fused shortcut add)
// ------------------------------------------------------------------
#define BM  128
#define BN  256
#define BK  32
#define LDK 40   // 64 B data + 16 B pad per row (matches TDM pad config)

__global__ __launch_bounds__(256)
void conv_gemm_kernel(const __bf16* __restrict__ A, const __bf16* __restrict__ in,
                      float* __restrict__ out, const float* __restrict__ bias,
                      const float* __restrict__ tproj, int Cin, int npos, int mode) {
  const int ncsh = (Cin == 256) ? 3 : 4;   // log2(Cin/32)
  const int nc   = 1 << ncsh;
  const int T    = npos << ncsh;           // K/32 k-steps
  const int K    = Cin * npos;
  const int n0   = blockIdx.x * BN;
  const int m0   = blockIdx.y * BM;
  const int tid  = threadIdx.x;
  const int lane = tid & 31;
  const int wv   = tid >> 5;
  const int wm   = wv & 1;    // M offset *64
  const int wn   = wv >> 1;   // N offset *64

  __shared__ __align__(16) __bf16 As[2][BM][LDK];
  __shared__ __align__(16) __bf16 Bs[2][BN][LDK];   // transposed: [n][k]

  v8f acc[4][4] = {};

  // B-tile mapping: thread -> one n column, all 32 k's of the slab
  const int nl   = tid;               // 0..255
  const int n    = n0 + nl;
  const int bimg = n >> 12;
  const int y    = (n >> 6) & 63;
  const int x    = n & 63;

  // wave-0 scalar gate (force s_cbranch: TDM ignores EXEC, so EXEC-masking
  // would issue it from every wave)
  const bool tdm_wave = (__builtin_amdgcn_readfirstlane(wv) == 0);

  auto issueA = [&](int t, int buf) {
    if (tdm_wave) {
      unsigned lds_byte = (unsigned)(size_t)(__attribute__((address_space(3))) void*)&As[buf][0][0];
      tdm_load_A(A + (size_t)m0 * K + (size_t)t * BK, lds_byte, K);
    }
  };

  auto gatherB = [&](int t, uint4 (&r)[4]) {
    const int p  = t >> ncsh;
    const int c0 = (t & (nc - 1)) << 5;
    const int dy = (npos == 1) ? 0 : (p / 3 - 1);
    const int dx = (npos == 1) ? 0 : (p - (p / 3) * 3 - 1);
    const int yy = y + dy;
    const int xx = x + dx;
    const bool inb = (yy >= 0) && (yy < 64) && (xx >= 0) && (xx < 64);
    union { __bf16 h[32]; uint4 v[4]; } u;
    if (inb) {
      const __bf16* srcb = in + (((bimg * Cin + c0) << 12) + (yy << 6) + xx);
      #pragma unroll
      for (int e = 0; e < 32; ++e) u.h[e] = srcb[e << 12];   // stride = HW
    } else {
      #pragma unroll
      for (int e = 0; e < 32; ++e) u.h[e] = (__bf16)0.0f;
    }
    #pragma unroll
    for (int q = 0; q < 4; ++q) r[q] = u.v[q];
  };

  auto storeB = [&](int buf, const uint4 (&r)[4]) {
    #pragma unroll
    for (int q = 0; q < 4; ++q) *(uint4*)&Bs[buf][nl][q * 8] = r[q];
  };

  auto computeTile = [&](int buf) {
    // 16-bit A/B fragment layout: lane -> M (or N) = lane&15; per-lane K
    // elements are two 8-element runs [cb,cb+8) and [cb+16,cb+24), cb=8*(lane>=16)
    const int lm = lane & 15;
    const int cb = (lane >> 4) << 3;
    union V16 { v16bf f; uint4 u[2]; };
    V16 bfv[4];
    #pragma unroll
    for (int j = 0; j < 4; ++j) {
      const __bf16* r = &Bs[buf][wn * 64 + j * 16 + lm][0];
      bfv[j].u[0] = *(const uint4*)(r + cb);
      bfv[j].u[1] = *(const uint4*)(r + cb + 16);
    }
    #pragma unroll
    for (int i = 0; i < 4; ++i) {
      V16 af;
      const __bf16* r = &As[buf][wm * 64 + i * 16 + lm][0];
      af.u[0] = *(const uint4*)(r + cb);
      af.u[1] = *(const uint4*)(r + cb + 16);
      #pragma unroll
      for (int j = 0; j < 4; ++j)
        acc[i][j] = __builtin_amdgcn_wmma_f32_16x16x32_bf16(
            false, af.f, false, bfv[j].f, (short)0, acc[i][j], false, false);
    }
  };

  // ---- prologue: stage k-step 0 into buffer 0 ----
  int buf = 0;
  {
    issueA(0, 0);
    uint4 r[4];
    gatherB(0, r);
    storeB(0, r);
    if (tdm_wave) tdm_wait0();
  }
  __syncthreads();

  // ---- pipelined main loop: stage t+1 while computing t ----
  for (int t = 0; t < T; ++t) {
    const int nb = buf ^ 1;
    const bool more = (t + 1 < T);
    uint4 r[4];
    if (more) {
      issueA(t + 1, nb);   // TDM overlaps the WMMAs below
      gatherB(t + 1, r);   // global-load latency overlaps too
    }
    computeTile(buf);
    if (more) {
      storeB(nb, r);
      if (tdm_wave) tdm_wait0();
    }
    __syncthreads();
    buf = nb;
  }

  // ---- epilogue: C/D layout element r -> M = r + 8*(lane>=16), N = lane&15 ----
  const int lm  = lane & 15;
  const int mhi = (lane >> 4) << 3;
  #pragma unroll
  for (int i = 0; i < 4; ++i) {
    #pragma unroll
    for (int j = 0; j < 4; ++j) {
      const int nn = n0 + wn * 64 + j * 16 + lm;
      const int bb = nn >> 12;
      const int sp = nn & 4095;
      #pragma unroll
      for (int r = 0; r < 8; ++r) {
        const int co   = m0 + wm * 64 + i * 16 + mhi + r;
        const int oidx = ((bb * COUT + co) << 12) + sp;
        float v = acc[i][j][r] + bias[co];
        if (mode == 0) v += tproj[bb * COUT + co];
        if (mode == 2) v += out[oidx];
        out[oidx] = v;
      }
    }
  }
}

// ------------------------------------------------------------------
extern "C" void kernel_launch(void* const* d_in, const int* in_sizes, int n_in,
                              void* d_out, int out_size, void* d_ws, size_t ws_size,
                              hipStream_t stream) {
  const float* x       = (const float*)d_in[0];
  const float* temb    = (const float*)d_in[1];
  const float* gn1_w   = (const float*)d_in[2];
  const float* gn1_b   = (const float*)d_in[3];
  const float* conv1_w = (const float*)d_in[4];
  const float* conv1_b = (const float*)d_in[5];
  const float* temb_w  = (const float*)d_in[6];
  const float* temb_b  = (const float*)d_in[7];
  const float* gn2_w   = (const float*)d_in[8];
  const float* gn2_b   = (const float*)d_in[9];
  const float* conv2_w = (const float*)d_in[10];
  const float* conv2_b = (const float*)d_in[11];
  const float* nin_w   = (const float*)d_in[12];
  const float* nin_b   = (const float*)d_in[13];
  float* out = (float*)d_out;

  char* ws = (char*)d_ws;
  size_t off = 0;
  auto take = [&](size_t bytes) {
    char* p = ws + off;
    off = (off + bytes + 255) & ~(size_t)255;
    return p;
  };
  __bf16* h1b   = (__bf16*)take((size_t)Bsz * CIN  * HW * 2);
  __bf16* xb    = (__bf16*)take((size_t)Bsz * CIN  * HW * 2);
  float*  h2    = (float*) take((size_t)Bsz * COUT * HW * 4);
  __bf16* h2s   = (__bf16*)take((size_t)Bsz * COUT * HW * 2);
  __bf16* w1p   = (__bf16*)take((size_t)COUT * CIN  * 9 * 2);
  __bf16* w2p   = (__bf16*)take((size_t)COUT * COUT * 9 * 2);
  __bf16* wnp   = (__bf16*)take((size_t)COUT * CIN * 2);
  float*  tpj   = (float*) take((size_t)Bsz * COUT * 4);
  float*  mean1 = (float*) take((size_t)Bsz * 32 * 4);
  float*  rstd1 = (float*) take((size_t)Bsz * 32 * 4);
  float*  mean2 = (float*) take((size_t)Bsz * 32 * 4);
  float*  rstd2 = (float*) take((size_t)Bsz * 32 * 4);

  // weight repack + temb projection (independent, cheap)
  pack_w3_kernel<<<(COUT * CIN  * 9 + 255) / 256, 256, 0, stream>>>(conv1_w, w1p, CIN);
  pack_w3_kernel<<<(COUT * COUT * 9 + 255) / 256, 256, 0, stream>>>(conv2_w, w2p, COUT);
  pack_w1_kernel<<<(COUT * CIN + 255) / 256, 256, 0, stream>>>(nin_w, wnp);
  temb_proj_kernel<<<(Bsz * COUT + 255) / 256, 256, 0, stream>>>(temb, temb_w, temb_b, tpj);

  // GN1 + SiLU -> bf16 (and bf16 copy of x)
  gn_stats_kernel<<<Bsz * 32, 256, 0, stream>>>(x, mean1, rstd1, CIN / 32);
  gn1_apply_kernel<<<(Bsz * CIN * HW) / 256, 256, 0, stream>>>(x, mean1, rstd1, gn1_w, gn1_b, h1b, xb);

  // conv1 (+bias +temb) -> h2 fp32
  conv_gemm_kernel<<<dim3(Bsz * HW / BN, COUT / BM), 256, 0, stream>>>(
      w1p, h1b, h2, conv1_b, tpj, CIN, 9, 0);

  // GN2 + SiLU -> bf16
  gn_stats_kernel<<<Bsz * 32, 256, 0, stream>>>(h2, mean2, rstd2, COUT / 32);
  gn2_apply_kernel<<<(Bsz * COUT * HW) / 256, 256, 0, stream>>>(h2, mean2, rstd2, gn2_w, gn2_b, h2s);

  // nin shortcut (1x1) writes d_out, then conv2 accumulates into it
  conv_gemm_kernel<<<dim3(Bsz * HW / BN, COUT / BM), 256, 0, stream>>>(
      wnp, xb, out, nin_b, nullptr, CIN, 1, 1);
  conv_gemm_kernel<<<dim3(Bsz * HW / BN, COUT / BM), 256, 0, stream>>>(
      w2p, h2s, out, conv2_b, nullptr, COUT, 9, 2);
}
